// PriorEnhancedMindEncoder_85650237817522
// MI455X (gfx1250) — compile-verified, matrix operand order checked by reading, not run on hardware
//
#include <hip/hip_runtime.h>
#include <hip/hip_bf16.h>

// ---------------------------------------------------------------------------
// PriorEnhancedMindEncoder on gfx1250 (MI455X).
// All matmuls go through v_wmma_f32_16x16x32_bf16 (bf16 A/B, f32 accumulate).
// Big GEMMs: 4-wave blocks computing 64x64 tiles, B K-panel double-buffered
// into LDS via GLOBAL_LOAD_ASYNC_TO_LDS_B128 (ASYNCcnt + barrier sync).
// Prior-MLP / adjacency normalization is provably dead (mask = adj>0 always
// true since exp(-x)*(1+sigmoid) > 0, and adj_norm feeds no output): skipped.
// ---------------------------------------------------------------------------

typedef __bf16 bf16;
typedef __attribute__((ext_vector_type(16))) __bf16       v16bf;
typedef __attribute__((ext_vector_type(8)))  float        v8f;
typedef __attribute__((ext_vector_type(4)))  unsigned int u32x4;
typedef int v4i __attribute__((vector_size(16)));

#define AS1 __attribute__((address_space(1)))
#define AS3 __attribute__((address_space(3)))

__device__ __forceinline__ bf16 f2bf(float f) {
    union { float f; unsigned u; } in; in.f = f;
    unsigned u = in.u + 0x7FFFu + ((in.u >> 16) & 1u);   // round-to-nearest-even
    union { unsigned short s; bf16 b; } out;
    out.s = (unsigned short)(u >> 16);
    return out.b;
}
__device__ __forceinline__ float gelu_exact(float x) {
    return 0.5f * x * (1.0f + erff(x * 0.7071067811865475f));
}

union ABFrag { u32x4 q[2]; v16bf v; };
union CFrag  { v8f v; float f[8]; };

// -------------------- CDNA5 async global->LDS helpers ----------------------
__device__ __forceinline__ void async_cp16(bf16* lds_dst, const bf16* gsrc) {
#if __has_builtin(__builtin_amdgcn_global_load_async_to_lds_b128)
    __builtin_amdgcn_global_load_async_to_lds_b128(
        (AS1 v4i*)gsrc, (AS3 v4i*)lds_dst, 0, 0);
#else
    unsigned loff = (unsigned)(unsigned long long)(AS3 void*)lds_dst;
    asm volatile("global_load_async_to_lds_b128 %0, %1, off"
                 :: "v"(loff), "v"(gsrc) : "memory");
#endif
}
__device__ __forceinline__ void wait_async_le0() {
#if __has_builtin(__builtin_amdgcn_s_wait_asynccnt)
    __builtin_amdgcn_s_wait_asynccnt(0);
#else
    asm volatile("s_wait_asynccnt 0x0" ::: "memory");
#endif
}
__device__ __forceinline__ void wait_async_le4() {
#if __has_builtin(__builtin_amdgcn_s_wait_asynccnt)
    __builtin_amdgcn_s_wait_asynccnt(4);
#else
    asm volatile("s_wait_asynccnt 0x4" ::: "memory");
#endif
}

// ---------------------------------------------------------------------------
// Tiled async WMMA GEMM: 128 threads (4 waves) per block, 64x64 C tile.
//   C[z] = act( A[z] (MxK bf16 row-major, K%64==0) *
//               Bt[z]^T (Bt: Nout x K bf16 row-major, Nout%64==0) + bias )
// B K-panels (64 cols x 64 k) double-buffered in LDS via async copies.
// grid = (ceil(M/64), Nout/64, Z).
// ---------------------------------------------------------------------------
#define KB   64
#define BPIT 72   // LDS row pitch in bf16 elems (pad 8 -> conflict-free b128)

__global__ __launch_bounds__(128)
void gemm_wmma_async(const bf16* __restrict__ A, const bf16* __restrict__ Bt,
                     const float* __restrict__ bias,
                     float* __restrict__ Cf, bf16* __restrict__ Cb,
                     int M, int K, int ldc, int act,
                     long long aZ, long long bZ, int Hh, long long cOuter, int headOff)
{
    __shared__ bf16 sB[2][64 * BPIT];

    const int z = blockIdx.z;
    const int tileM = blockIdx.x, tileN = blockIdx.y;
    const int tid = threadIdx.x;
    const int wave = tid >> 5, lane = tid & 31;
    const int half = lane >> 4, l16 = lane & 15;

    A  += (long long)z * aZ;
    Bt += (long long)z * bZ;
    const long long cbase = (long long)(z / Hh) * cOuter + (long long)(z % Hh) * headOff;

    int arow = tileM * 64 + wave * 16 + l16;
    if (arow >= M) arow = M - 1;                 // clamp keeps EXEC uniform at WMMA
    const bf16* ap = A + (long long)arow * K + half * 8;

    const int c0 = tileN * 64;
    const int nchunk = K / KB;

    // stage one 64col x 64k B panel: each thread copies 4 x 16B (async)
    auto stage = [&](int chunk, int buf) {
        const bf16* gbase = Bt + (long long)c0 * K + chunk * KB;
#pragma unroll
        for (int i = 0; i < 4; ++i) {
            const int idx = tid + i * 128;       // 0..511
            const int col = idx >> 3, seg = idx & 7;
            async_cp16(&sB[buf][col * BPIT + seg * 8],
                       gbase + (long long)col * K + seg * 8);
        }
    };

    v8f acc[4];
#pragma unroll
    for (int nf = 0; nf < 4; ++nf) acc[nf] = (v8f){0.f,0.f,0.f,0.f,0.f,0.f,0.f,0.f};

    stage(0, 0);
    for (int ch = 0; ch < nchunk; ++ch) {
        if (ch + 1 < nchunk) { stage(ch + 1, (ch + 1) & 1); wait_async_le4(); }
        else                 { wait_async_le0(); }
        __syncthreads();                          // panel `ch` visible block-wide

        const bf16* sb = &sB[ch & 1][0];
        const int kg = ch * KB;
        __builtin_prefetch(ap + kg + KB, 0, 3);   // next A chunk to near caches
#pragma unroll
        for (int s = 0; s < 2; ++s) {
            ABFrag a;
            a.q[0] = *(const u32x4*)(ap + kg + s * 32);
            a.q[1] = *(const u32x4*)(ap + kg + s * 32 + 16);
#pragma unroll
            for (int nf = 0; nf < 4; ++nf) {
                ABFrag b;
                const bf16* bp = sb + (nf * 16 + l16) * BPIT + s * 32 + half * 8;
                b.q[0] = *(const u32x4*)(bp);
                b.q[1] = *(const u32x4*)(bp + 16);
                acc[nf] = __builtin_amdgcn_wmma_f32_16x16x32_bf16(
                    false, a.v, false, b.v, (short)0, acc[nf], false, false);
            }
        }
        __syncthreads();                          // done reading before overwrite
    }

#pragma unroll
    for (int nf = 0; nf < 4; ++nf) {
        const int col = c0 + nf * 16 + l16;
        const float bv = bias ? bias[col] : 0.0f;
        CFrag res; res.v = acc[nf];
#pragma unroll
        for (int g = 0; g < 8; ++g) {
            const int orow = tileM * 64 + wave * 16 + half * 8 + g;
            if (orow < M) {
                float v = res.f[g] + bv;
                if (act == 1) v = gelu_exact(v);
                const long long idx = cbase + (long long)orow * ldc + col;
                if (Cf) Cf[idx] = v;
                if (Cb) Cb[idx] = f2bf(v);
            }
        }
    }
}

// ---------------------------------------------------------------------------
// Single-tile WMMA GEMM (used for the Nout=32 head-aggregation GEMMs).
// ---------------------------------------------------------------------------
__global__ __launch_bounds__(32)
void gemm_wmma(const bf16* __restrict__ A, const bf16* __restrict__ Bt,
               const float* __restrict__ bias,
               float* __restrict__ Cf, bf16* __restrict__ Cb,
               int M, int K, int ldc, int act,
               long long aZ, long long bZ, int Hh, long long cOuter, int headOff)
{
    const int z     = blockIdx.z;
    const int tileM = blockIdx.x;
    const int tileN = blockIdx.y;
    const int lane  = threadIdx.x & 31;
    const int half  = lane >> 4;
    const int l16   = lane & 15;

    A  += (long long)z * aZ;
    Bt += (long long)z * bZ;
    const long long cbase = (long long)(z / Hh) * cOuter + (long long)(z % Hh) * headOff;

    int arow = tileM * 16 + l16;
    if (arow >= M) arow = M - 1;
    const int bcol = tileN * 16 + l16;

    const bf16* ap = A  + (long long)arow * K + half * 8;
    const bf16* bp = Bt + (long long)bcol * K + half * 8;

    v8f acc = {0.f, 0.f, 0.f, 0.f, 0.f, 0.f, 0.f, 0.f};
    for (int k = 0; k < K; k += 32) {
        ABFrag a, b;
        a.q[0] = *(const u32x4*)(ap + k);
        a.q[1] = *(const u32x4*)(ap + k + 16);
        b.q[0] = *(const u32x4*)(bp + k);
        b.q[1] = *(const u32x4*)(bp + k + 16);
        __builtin_prefetch(ap + k + 64, 0, 3);
        __builtin_prefetch(bp + k + 64, 0, 3);
        acc = __builtin_amdgcn_wmma_f32_16x16x32_bf16(
            false, a.v, false, b.v, (short)0, acc, false, false);
    }

    const float bv = bias ? bias[bcol] : 0.0f;
    CFrag res; res.v = acc;
#pragma unroll
    for (int g = 0; g < 8; ++g) {
        const int orow = tileM * 16 + half * 8 + g;
        if (orow < M) {
            float v = res.f[g] + bv;
            if (act == 1) v = gelu_exact(v);
            const long long idx = cbase + (long long)orow * ldc + tileN * 16 + l16;
            if (Cf) Cf[idx] = v;
            if (Cb) Cb[idx] = f2bf(v);
        }
    }
}

// ---------------------------------------------------------------------------
// fp32 [R x C] (row pitch `pitch`, batched by (b,h)) -> bf16 [C x Rpad], pad=0
// ---------------------------------------------------------------------------
__global__ void transpose_pad(const float* __restrict__ src, bf16* __restrict__ dst,
                              int R, int C, int Rpad, int pitch,
                              long long srcOuter, int Hh, long long srcHead)
{
    const int z = blockIdx.z;
    const float* s = src + (long long)(z / Hh) * srcOuter + (long long)(z % Hh) * srcHead;
    bf16* d = dst + (long long)z * C * Rpad;
    const int total = C * Rpad;
    for (int i = blockIdx.x * blockDim.x + threadIdx.x; i < total; i += gridDim.x * blockDim.x) {
        const int c = i / Rpad, r = i % Rpad;
        d[i] = (r < R) ? f2bf(s[(long long)r * pitch + c]) : f2bf(0.0f);
    }
}

__global__ void pad_convert(const float* __restrict__ src, bf16* __restrict__ dst,
                            int C, int Cpad, long long total)
{
    for (long long i = blockIdx.x * (long long)blockDim.x + threadIdx.x; i < total;
         i += (long long)gridDim.x * blockDim.x) {
        const long long r = i / Cpad; const int k = (int)(i % Cpad);
        dst[i] = (k < C) ? f2bf(src[r * C + k]) : f2bf(0.0f);
    }
}

__global__ void concat_bf(const float* __restrict__ a, const float* __restrict__ b,
                          bf16* __restrict__ dst, long long total)
{
    for (long long i = blockIdx.x * (long long)blockDim.x + threadIdx.x; i < total;
         i += (long long)gridDim.x * blockDim.x) {
        const long long r = i >> 9; const int c = (int)(i & 511);
        dst[i] = f2bf(c < 256 ? a[r * 256 + c] : b[r * 256 + (c - 256)]);
    }
}

__global__ __launch_bounds__(256)
void ln_gelu_k(const float* __restrict__ x, const float* __restrict__ g,
               const float* __restrict__ b, bf16* __restrict__ out)
{
    __shared__ float red[256]; __shared__ float stat;
    const int row = blockIdx.x, tid = threadIdx.x;
    const float v = x[(long long)row * 256 + tid];
    red[tid] = v; __syncthreads();
    for (int s = 128; s; s >>= 1) { if (tid < s) red[tid] += red[tid + s]; __syncthreads(); }
    if (tid == 0) stat = red[0] * (1.0f / 256.0f);
    __syncthreads();
    const float m = stat, d = v - m;
    red[tid] = d * d; __syncthreads();
    for (int s = 128; s; s >>= 1) { if (tid < s) red[tid] += red[tid + s]; __syncthreads(); }
    if (tid == 0) stat = red[0] * (1.0f / 256.0f);
    __syncthreads();
    const float y = d * rsqrtf(stat + 1e-5f) * g[tid] + b[tid];
    out[(long long)row * 256 + tid] = f2bf(gelu_exact(y));
}

__global__ __launch_bounds__(256)
void gat_scores(const float* __restrict__ ht, const float* __restrict__ a,
                float* __restrict__ ei, float* __restrict__ ej, int Hh, int dh, int Nn)
{
    const int rn = blockIdx.x;                 // b*Nn + n
    const int h = threadIdx.x >> 5, lane = threadIdx.x & 31;
    const float* base = ht + (long long)rn * Hh * dh + h * dh;
    const float* av = a + (long long)h * 2 * dh;
    float si = 0.f, sj = 0.f;
    for (int d = lane; d < dh; d += 32) {
        const float v = base[d];
        si += v * av[d];
        sj += v * av[dh + d];
    }
    for (int off = 16; off; off >>= 1) { si += __shfl_down(si, off); sj += __shfl_down(sj, off); }
    if (lane == 0) {
        const int b = rn / Nn, n = rn % Nn;
        ei[((long long)b * Hh + h) * Nn + n] = si;
        ej[((long long)b * Hh + h) * Nn + n] = sj;
    }
}

__global__ __launch_bounds__(128)
void gat_alpha(const float* __restrict__ ei, const float* __restrict__ ej,
               bf16* __restrict__ alpha, int Nn, int Kp)
{
    __shared__ float sej[384]; __shared__ float red[128];
    const int z = blockIdx.y, n = blockIdx.x, tid = threadIdx.x;
    for (int m = tid; m < Kp; m += 128) sej[m] = (m < Nn) ? ej[(long long)z * Nn + m] : 0.f;
    __syncthreads();
    const float myei = ei[(long long)z * Nn + n];
    float mx = -1e30f;
    for (int m = tid; m < Nn; m += 128) {
        float e = myei + sej[m]; e = e > 0.f ? e : 0.2f * e;
        mx = fmaxf(mx, e);
    }
    red[tid] = mx; __syncthreads();
    for (int s = 64; s; s >>= 1) { if (tid < s) red[tid] = fmaxf(red[tid], red[tid + s]); __syncthreads(); }
    mx = red[0]; __syncthreads();
    float sum = 0.f;
    for (int m = tid; m < Nn; m += 128) {
        float e = myei + sej[m]; e = e > 0.f ? e : 0.2f * e;
        sum += expf(e - mx);
    }
    red[tid] = sum; __syncthreads();
    for (int s = 64; s; s >>= 1) { if (tid < s) red[tid] += red[tid + s]; __syncthreads(); }
    const float rinv = 1.0f / red[0];
    bf16* row = alpha + (long long)z * Nn * Kp + (long long)n * Kp;
    for (int m = tid; m < Kp; m += 128) {
        float v = 0.f;
        if (m < Nn) { float e = myei + sej[m]; e = e > 0.f ? e : 0.2f * e; v = expf(e - mx) * rinv; }
        row[m] = f2bf(v);
    }
}

__global__ __launch_bounds__(256)
void row_norm(const float* __restrict__ h, float* __restrict__ norms)
{
    __shared__ float red[256];
    const int row = blockIdx.x, tid = threadIdx.x;
    const float v = h[(long long)row * 256 + tid];
    red[tid] = v * v; __syncthreads();
    for (int s = 128; s; s >>= 1) { if (tid < s) red[tid] += red[tid + s]; __syncthreads(); }
    if (tid == 0) norms[row] = sqrtf(red[0]);
}

__global__ __launch_bounds__(512)
void softmax_nodes(const float* __restrict__ norms, float* __restrict__ attn, int Nn)
{
    __shared__ float red[512];
    const int b = blockIdx.x, tid = threadIdx.x;
    const float v = (tid < Nn) ? norms[(long long)b * Nn + tid] : -1e30f;
    red[tid] = v; __syncthreads();
    for (int s = 256; s; s >>= 1) { if (tid < s) red[tid] = fmaxf(red[tid], red[tid + s]); __syncthreads(); }
    const float mx = red[0]; __syncthreads();
    const float e = (tid < Nn) ? expf(v - mx) : 0.f;
    red[tid] = e; __syncthreads();
    for (int s = 256; s; s >>= 1) { if (tid < s) red[tid] += red[tid + s]; __syncthreads(); }
    if (tid < Nn) attn[(long long)b * Nn + tid] = e / red[0];
}

__global__ __launch_bounds__(256)
void readout_ge(const float* __restrict__ attn, const float* __restrict__ h,
                bf16* __restrict__ geB, int Nn, int D)
{
    const int b = blockIdx.x, d = threadIdx.x;
    float acc = 0.f;
    for (int n = 0; n < Nn; ++n)
        acc += attn[(long long)b * Nn + n] * h[((long long)b * Nn + n) * D + d];
    geB[(long long)b * D + d] = f2bf(acc);
}

// ---------------------------------------------------------------------------
extern "C" void kernel_launch(void* const* d_in, const int* in_sizes, int n_in,
                              void* d_out, int out_size, void* d_ws, size_t ws_size,
                              hipStream_t stream)
{
    (void)in_sizes; (void)n_in; (void)out_size; (void)ws_size;
    const float* mind = (const float*)d_in[0];
    const float* roi  = (const float*)d_in[1];
    // d_in[2] (group_prior) and d_in[9..12] (prior MLP) are dead: mask adj>0 is always true.
    const float* Wnp = (const float*)d_in[3];  const float* bnp = (const float*)d_in[4];
    const float* Wrf = (const float*)d_in[5];  const float* brf = (const float*)d_in[6];
    const float* lng = (const float*)d_in[7];  const float* lnb = (const float*)d_in[8];
    const float* gW0 = (const float*)d_in[13]; const float* ga0 = (const float*)d_in[14];
    const float* gW1 = (const float*)d_in[15]; const float* ga1 = (const float*)d_in[16];
    const float* gW2 = (const float*)d_in[17]; const float* ga2 = (const float*)d_in[18];
    const float* pW  = (const float*)d_in[19]; const float* pb  = (const float*)d_in[20];
    const float* Wr1 = (const float*)d_in[21]; const float* br1 = (const float*)d_in[22];
    const float* Wr2 = (const float*)d_in[23]; const float* br2 = (const float*)d_in[24];

    constexpr int B = 64, Nn = 360, DH = 256, H = 8;
    constexpr int BN = B * Nn;           // 23040 (multiple of 64)
    constexpr int NP = 384;              // node dim padded (mult of 64)

    float* out    = (float*)d_out;
    float* out_ge = out;
    float* out_h  = out + (long long)B * DH;
    float* out_at = out_h + (long long)BN * DH;

    size_t off = 0;
    auto alloc = [&](size_t bytes) -> char* {
        char* p = (char*)d_ws + off;
        off += (bytes + 255) & ~(size_t)255;
        return p;
    };

    bf16* wnpT = (bf16*)alloc((size_t)256 * 384 * 2);
    bf16* wrfT = (bf16*)alloc((size_t)256 * 512 * 2);
    bf16* g0T  = (bf16*)alloc((size_t)256 * 256 * 2);
    bf16* g1T  = (bf16*)alloc((size_t)256 * 256 * 2);
    bf16* g2T  = (bf16*)alloc((size_t)2048 * 256 * 2);
    bf16* pT   = (bf16*)alloc((size_t)256 * 2048 * 2);
    bf16* r1T  = (bf16*)alloc((size_t)256 * 256 * 2);
    bf16* r2T  = (bf16*)alloc((size_t)256 * 256 * 2);

    bf16*  mind_bf = (bf16*)alloc((size_t)BN * NP * 2);
    float* nm_f    = (float*)alloc((size_t)BN * 256 * 4);
    bf16*  cat_bf  = (bf16*)alloc((size_t)BN * 512 * 2);
    float* nf_f    = (float*)alloc((size_t)BN * 256 * 4);
    bf16*  hA      = (bf16*)alloc((size_t)BN * 256 * 2);
    bf16*  hB      = (bf16*)alloc((size_t)BN * 256 * 2);
    float* ht_f    = (float*)alloc((size_t)BN * 2048 * 4);
    float* eib     = (float*)alloc((size_t)B * H * Nn * 4);
    float* ejb     = (float*)alloc((size_t)B * H * Nn * 4);
    bf16*  alphaB  = (bf16*)alloc((size_t)B * H * Nn * NP * 2);
    bf16*  htT     = (bf16*)alloc((size_t)B * H * 256 * NP * 2);
    bf16*  hagg_bf = (bf16*)alloc((size_t)BN * 2048 * 2);
    float* norms   = (float*)alloc((size_t)BN * 4);
    bf16*  ge_bf   = (bf16*)alloc((size_t)B * 256 * 2);
    bf16*  t_bf    = (bf16*)alloc((size_t)B * 256 * 2);

    // 64x64-tile async kernel (K%64==0, Nout%64==0)
    auto GA = [&](const bf16* A, const bf16* Bt, const float* bias,
                  float* Cf, bf16* Cb, int M, int K, int Nout, int ldc, int act,
                  long long aZ, long long bZ, int Hh, long long cOuter, int headOff, int Z) {
        dim3 grid((M + 63) / 64, Nout / 64, Z);
        gemm_wmma_async<<<grid, 128, 0, stream>>>(A, Bt, bias, Cf, Cb, M, K, ldc, act,
                                                  aZ, bZ, Hh, cOuter, headOff);
    };
    // single-tile kernel (small Nout)
    auto G = [&](const bf16* A, const bf16* Bt, const float* bias,
                 float* Cf, bf16* Cb, int M, int K, int Nout, int ldc, int act,
                 long long aZ, long long bZ, int Hh, long long cOuter, int headOff, int Z) {
        dim3 grid((M + 15) / 16, Nout / 16, Z);
        gemm_wmma<<<grid, 32, 0, stream>>>(A, Bt, bias, Cf, Cb, M, K, ldc, act,
                                           aZ, bZ, Hh, cOuter, headOff);
    };
    auto T = [&](const float* src, bf16* dst, int R, int C, int Rpad, int pitch) {
        dim3 grid(((size_t)C * Rpad + 255) / 256, 1, 1);
        transpose_pad<<<grid, 256, 0, stream>>>(src, dst, R, C, Rpad, pitch, 0LL, 1, 0LL);
    };

    // weight transposes (Bt operands: [Nout x K] bf16, K zero-padded)
    T(Wnp, wnpT, 360, 256, 384, 256);
    T(Wrf, wrfT, 512, 256, 512, 256);
    T(gW0, g0T,  256, 256, 256, 256);
    T(gW1, g1T,  256, 256, 256, 256);
    T(gW2, g2T,  256, 2048, 256, 2048);
    T(pW,  pT,   2048, 256, 2048, 256);
    T(Wr1, r1T,  256, 256, 256, 256);
    T(Wr2, r2T,  256, 256, 256, 256);

    // node feature fusion
    pad_convert<<<((long long)BN * NP + 255) / 256, 256, 0, stream>>>(mind, mind_bf, 360, NP, (long long)BN * NP);
    GA(mind_bf, wnpT, bnp, nm_f, nullptr, BN, NP, 256, 256, 0, 0, 0, 1, 0, 0, 1);
    concat_bf<<<((long long)BN * 512 + 255) / 256, 256, 0, stream>>>(nm_f, roi, cat_bf, (long long)BN * 512);
    GA(cat_bf, wrfT, brf, nf_f, nullptr, BN, 512, 256, 256, 0, 0, 0, 1, 0, 0, 1);
    ln_gelu_k<<<BN, 256, 0, stream>>>(nf_f, lng, lnb, hA);

    // hierarchical GAT
    auto gat = [&](const bf16* hin, const bf16* WT, const float* avec, int dh, bf16* outB) {
        const int HD = H * dh;
        GA(hin, WT, nullptr, ht_f, nullptr, BN, 256, HD, HD, 0, 0, 0, 1, 0, 0, 1);
        gat_scores<<<BN, 256, 0, stream>>>(ht_f, avec, eib, ejb, H, dh, Nn);
        gat_alpha<<<dim3(Nn, B * H), 128, 0, stream>>>(eib, ejb, alphaB, Nn, NP);
        {   // ht[b,:,h,:] -> htT[z=(b*H+h)][dh][NP] bf16 (zero-pad rows)
            dim3 grid(((size_t)dh * NP + 255) / 256, 1, B * H);
            transpose_pad<<<grid, 256, 0, stream>>>(ht_f, htT, Nn, dh, NP, HD,
                                                    (long long)Nn * HD, H, (long long)dh);
        }
        // hagg[b,n,h*dh+d] = alpha[z] @ htT[z]^T  (batched WMMA)
        if (dh >= 64) {
            GA(alphaB, htT, nullptr, nullptr, outB, Nn, NP, dh, HD, 0,
               (long long)Nn * NP, (long long)dh * NP, H, (long long)Nn * HD, dh, B * H);
        } else {
            G(alphaB, htT, nullptr, nullptr, outB, Nn, NP, dh, HD, 0,
              (long long)Nn * NP, (long long)dh * NP, H, (long long)Nn * HD, dh, B * H);
        }
    };
    gat(hA, g0T, ga0, 32, hB);
    gat(hB, g1T, ga1, 32, hA);
    gat(hA, g2T, ga2, 256, hagg_bf);
    // concat=False layer projection -> final h, straight into d_out
    GA(hagg_bf, pT, pb, out_h, nullptr, BN, 2048, 256, 256, 0, 0, 0, 1, 0, 0, 1);

    // attention readout
    row_norm<<<BN, 256, 0, stream>>>(out_h, norms);
    softmax_nodes<<<B, 512, 0, stream>>>(norms, out_at, Nn);
    readout_ge<<<B, 256, 0, stream>>>(out_at, out_h, ge_bf, Nn, DH);
    GA(ge_bf, r1T, br1, nullptr, t_bf, B, 256, 256, 256, /*gelu*/1, 0, 0, 1, 0, 0, 1);
    GA(t_bf,  r2T, br2, out_ge, nullptr, B, 256, 256, 256, 0, 0, 0, 1, 0, 0, 1);
}
